// GNNMultiEdgeset_56169582297599
// MI455X (gfx1250) — matrix operand order repaired and checked by reading, not compiled, not run on hardware
//
#include <hip/hip_runtime.h>

#define NN   20000
#define NE   40000
#define NG   1000
#define RCN  8        // R*C virtual copies
#define HID  128
#define INF  64
#define NL   3

typedef __attribute__((ext_vector_type(16))) _Float16 v16h;
typedef __attribute__((ext_vector_type(8)))  _Float16 v8h;
typedef __attribute__((ext_vector_type(8)))  float    v8f;

__device__ __forceinline__ float gelu_erf(float v) {
    return 0.5f * v * (1.0f + erff(v * 0.70710678118654752440f));
}

// Pack 16 f32 (already gathered per the 16-bit A-matrix lane swizzle) into a v16h.
__device__ __forceinline__ v16h cvt_afrag(float4 q0, float4 q1, float4 q2, float4 q3) {
    v16h a;
    a[0]=(_Float16)q0.x;  a[1]=(_Float16)q0.y;  a[2]=(_Float16)q0.z;  a[3]=(_Float16)q0.w;
    a[4]=(_Float16)q1.x;  a[5]=(_Float16)q1.y;  a[6]=(_Float16)q1.z;  a[7]=(_Float16)q1.w;
    a[8]=(_Float16)q2.x;  a[9]=(_Float16)q2.y;  a[10]=(_Float16)q2.z; a[11]=(_Float16)q2.w;
    a[12]=(_Float16)q3.x; a[13]=(_Float16)q3.y; a[14]=(_Float16)q3.z; a[15]=(_Float16)q3.w;
    return a;
}

// B fragment from fragment-major swizzled LDS weights: one contiguous 32B v16h load.
__device__ __forceinline__ v16h load_bfrag(const _Float16* sW, int kc, int nt, int lane) {
    return *(const v16h*)(sW + (((kc * 8 + nt) * 32 + lane) << 4));
}

// Swizzle f32 KxN (N=128) weights into f16 B-fragment-major layout:
//   Wz[((kc*8 + nt)*32 + lane)*16 + h] = W[(kc*32 + (lane>>4)*8 + (h&7) + (h>>3)*16)*128 + nt*16 + (lane&15)]
__global__ void swizzle_w16_kernel(const float* __restrict__ W, _Float16* __restrict__ Wz, int K) {
    int i = blockIdx.x * blockDim.x + threadIdx.x;
    if (i >= K * HID) return;
    int h    = i & 15;
    int lane = (i >> 4) & 31;
    int nt   = (i >> 9) & 7;
    int kc   = i >> 12;
    int k = kc * 32 + ((lane >> 4) << 3) + (h & 7) + ((h >> 3) << 4);
    int n = nt * 16 + (lane & 15);
    Wz[i] = (_Float16)W[k * HID + n];
}

// x0 = x_feat @ W_atom + b_atom, replicated into all RCN copies.  M=20000, K=64, N=128.
__global__ void __launch_bounds__(256)
atom_encoder_kernel(const float* __restrict__ X,       // [NN, 64]
                    const _Float16* __restrict__ Wg,   // [64*128] f16, swizzled (2 kc)
                    const float* __restrict__ bias,    // [128]
                    float* __restrict__ Out)           // [RCN, NN, 128]
{
    __shared__ __align__(32) _Float16 sW[INF * HID];   // 16 KB
    int t = threadIdx.x;
    {
        const uint4* g = (const uint4*)Wg;
        uint4* s = (uint4*)sW;
        for (int i = t; i < INF * HID / 8; i += 256) s[i] = g[i];
    }
    __syncthreads();

    int wave = t >> 5, lane = t & 31;
    long tile = (long)blockIdx.x * 8 + wave;
    if (tile * 16 >= NN) return;            // no barriers after this point
    long rowBase = tile * 16;

    int mrowA = lane & 15;
    int hiA   = (lane >> 4) << 3;           // 0 or 8
    int nA    = lane & 15;

    const float* arow = X + (rowBase + mrowA) * INF;
    v16h afrag[2];
#pragma unroll
    for (int kc = 0; kc < 2; ++kc) {
        const float* p = arow + kc * 32 + hiA;
        afrag[kc] = cvt_afrag(*(const float4*)(p),      *(const float4*)(p + 4),
                              *(const float4*)(p + 16), *(const float4*)(p + 20));
    }

    v8f acc[8];
#pragma unroll
    for (int nt = 0; nt < 8; ++nt) {
        float bv = bias[nt * 16 + nA];
#pragma unroll
        for (int v = 0; v < 8; ++v) acc[nt][v] = bv;
    }
#pragma unroll
    for (int kc = 0; kc < 2; ++kc)
#pragma unroll
        for (int nt = 0; nt < 8; ++nt)
            acc[nt] = __builtin_amdgcn_wmma_f32_16x16x32_f16(
                false, afrag[kc], false, load_bfrag(sW, kc, nt, lane),
                (short)0, acc[nt], false, false);

#pragma unroll
    for (int nt = 0; nt < 8; ++nt)
#pragma unroll
        for (int v = 0; v < 8; ++v) {
            float val = acc[nt][v];
            long coloff = (rowBase + hiA + v) * HID + nt * 16 + nA;
#pragma unroll
            for (int rc = 0; rc < RCN; ++rc)
                Out[(long)rc * NN * HID + coloff] = val;
        }
}

// Fused conv MLP:  Out = [gelu?]( gelu(A@W1+b1) @ W2 + b2 ),  K=N=128.
// 8 waves/WG, one 16-row M-tile per wave; swizzled f16 W1/W2 staged in LDS;
// GEMM1 result re-laid-out through LDS (reusing the sW1 region) into A fragments.
__global__ void __launch_bounds__(256)
fused_mlp_kernel(const float* __restrict__ A,      // [M, 128] f32
                 const _Float16* __restrict__ W1,  // [128*128] f16 swizzled
                 const float* __restrict__ b1,
                 const _Float16* __restrict__ W2,  // [128*128] f16 swizzled
                 const float* __restrict__ b2,
                 float* __restrict__ Out,          // [M, 128] f32
                 int M, int gelu_out)
{
    __shared__ __align__(32) _Float16 sW1[HID * HID];   // 32 KB, reused as intermediate tiles
    __shared__ __align__(32) _Float16 sW2[HID * HID];   // 32 KB
    int t = threadIdx.x;
    {
        const uint4* g1 = (const uint4*)W1;
        const uint4* g2 = (const uint4*)W2;
        uint4* s1 = (uint4*)sW1;
        uint4* s2 = (uint4*)sW2;
        for (int i = t; i < HID * HID / 8; i += 256) { s1[i] = g1[i]; s2[i] = g2[i]; }
    }
    __syncthreads();

    int wave = t >> 5, lane = t & 31;
    long tile = (long)blockIdx.x * 8 + wave;
    bool active = (tile * 16) < (long)M;
    long rowBase = tile * 16;

    int mrowA = lane & 15;
    int hiA   = (lane >> 4) << 3;   // 0/8
    int nA    = lane & 15;

    v8f acc[8];
    if (active) {
        const float* arow = A + (rowBase + mrowA) * HID;
        v16h afrag[4];
#pragma unroll
        for (int kc = 0; kc < 4; ++kc) {
            const float* p = arow + kc * 32 + hiA;
            afrag[kc] = cvt_afrag(*(const float4*)(p),      *(const float4*)(p + 4),
                                  *(const float4*)(p + 16), *(const float4*)(p + 20));
        }
#pragma unroll
        for (int nt = 0; nt < 8; ++nt) {
            float bv = b1[nt * 16 + nA];
#pragma unroll
            for (int v = 0; v < 8; ++v) acc[nt][v] = bv;
        }
#pragma unroll
        for (int kc = 0; kc < 4; ++kc)
#pragma unroll
            for (int nt = 0; nt < 8; ++nt)
                acc[nt] = __builtin_amdgcn_wmma_f32_16x16x32_f16(
                    false, afrag[kc], false, load_bfrag(sW1, kc, nt, lane),
                    (short)0, acc[nt], false, false);
    }

    __syncthreads();                        // sW1 fully consumed -> reuse as per-wave tile
    _Float16* sT = sW1 + wave * (16 * HID); // 16x128 f16 per wave

    if (active) {
        // D layout -> row-major f16 tile (with gelu)
#pragma unroll
        for (int nt = 0; nt < 8; ++nt)
#pragma unroll
            for (int v = 0; v < 8; ++v)
                sT[(hiA + v) * HID + nt * 16 + nA] =
                    (_Float16)gelu_erf(acc[nt][v]);

        // Re-load as A fragments (same-wave LDS ops are in order).
        v16h a2[4];
#pragma unroll
        for (int kc = 0; kc < 4; ++kc) {
            const _Float16* base = sT + mrowA * HID + kc * 32 + hiA;
            v8h lo = *(const v8h*)(base);
            v8h hi = *(const v8h*)(base + 16);
            v16h a;
#pragma unroll
            for (int h = 0; h < 8; ++h) { a[h] = lo[h]; a[8 + h] = hi[h]; }
            a2[kc] = a;
        }
#pragma unroll
        for (int nt = 0; nt < 8; ++nt) {
            float bv = b2[nt * 16 + nA];
#pragma unroll
            for (int v = 0; v < 8; ++v) acc[nt][v] = bv;
        }
#pragma unroll
        for (int kc = 0; kc < 4; ++kc)
#pragma unroll
            for (int nt = 0; nt < 8; ++nt)
                acc[nt] = __builtin_amdgcn_wmma_f32_16x16x32_f16(
                    false, a2[kc], false, load_bfrag(sW2, kc, nt, lane),
                    (short)0, acc[nt], false, false);

#pragma unroll
        for (int nt = 0; nt < 8; ++nt)
#pragma unroll
            for (int v = 0; v < 8; ++v) {
                float val = acc[nt][v];
                if (gelu_out) val = gelu_erf(val);
                Out[(rowBase + hiA + v) * HID + nt * 16 + nA] = val;
            }
    }
}

// edge_emb = edge_attr @ W_bond[l] + b_bond[l]   (K=16: trivial VALU GEMM)
__global__ void edge_emb_kernel(const float* __restrict__ EA,  // [NE,16]
                                const float* __restrict__ Wb,  // [16,128]
                                const float* __restrict__ bb,  // [128]
                                float* __restrict__ EE)        // [NE,128]
{
    int i = blockIdx.x * blockDim.x + threadIdx.x;
    if (i >= NE * HID) return;
    int e = i >> 7, f = i & 127;
    const float* a = EA + e * 16;
    float s = bb[f];
#pragma unroll
    for (int k = 0; k < 16; ++k) s += a[k] * Wb[k * HID + f];
    EE[i] = s;
}

// agg := (1+eps[l]) * x   (scatter target initializer; fuses the GIN eps term)
__global__ void init_agg_kernel(const float* __restrict__ X, const float* __restrict__ eps,
                                int l, float* __restrict__ Agg)
{
    long i = (long)blockIdx.x * blockDim.x + threadIdx.x;
    if (i >= (long)RCN * NN * HID / 4) return;
    float sc = 1.0f + eps[l];
    float4 v = ((const float4*)X)[i];
    v.x *= sc; v.y *= sc; v.z *= sc; v.w *= sc;
    ((float4*)Agg)[i] = v;
}

// One wave per (copy, edge): msg = gelu(x[src]+ee)*w -> atomic scatter to dst.
__global__ void __launch_bounds__(256)
msg_scatter_kernel(const float* __restrict__ X,    // [RCN,NN,128]
                   const float* __restrict__ EE,   // [NE,128]
                   const float* __restrict__ EW,   // [RCN,NE]
                   const int* __restrict__ srcI, const int* __restrict__ dstI,
                   float* __restrict__ Agg)        // [RCN,NN,128]
{
    int gw = blockIdx.x * 8 + (threadIdx.x >> 5);
    int lane = threadIdx.x & 31;
    if (gw >= RCN * NE) return;
    int rc = gw / NE;
    int e  = gw - rc * NE;
    int s = srcI[e], d = dstI[e];
    float w = EW[(long)rc * NE + e];
    float4 xs = *((const float4*)(X + ((long)rc * NN + s) * HID) + lane);
    float4 ee = *((const float4*)(EE + (long)e * HID) + lane);
    float4 m;
    m.x = gelu_erf(xs.x + ee.x) * w;
    m.y = gelu_erf(xs.y + ee.y) * w;
    m.z = gelu_erf(xs.z + ee.z) * w;
    m.w = gelu_erf(xs.w + ee.w) * w;
    float* out = Agg + ((long)rc * NN + d) * HID + lane * 4;
    unsafeAtomicAdd(out + 0, m.x);
    unsafeAtomicAdd(out + 1, m.y);
    unsafeAtomicAdd(out + 2, m.z);
    unsafeAtomicAdd(out + 3, m.w);
}

__global__ void zero_kernel(float* __restrict__ p, long n) {
    long i = (long)blockIdx.x * blockDim.x + threadIdx.x;
    if (i < n) p[i] = 0.0f;
}

// One wave per (copy, node): masked feature scatter into per-graph sums.
__global__ void __launch_bounds__(256)
pool_scatter_kernel(const float* __restrict__ Y,     // [RCN,NN,128]
                    const float* __restrict__ mask,  // [RCN,NN]
                    const int* __restrict__ batch,   // [NN]
                    float* __restrict__ num,         // [RCN,NG,128]
                    float* __restrict__ den)         // [RCN,NG]
{
    int gw = blockIdx.x * 8 + (threadIdx.x >> 5);
    int lane = threadIdx.x & 31;
    if (gw >= RCN * NN) return;
    int rc = gw / NN;
    int n  = gw - rc * NN;
    int g  = batch[n];
    float m = mask[(long)rc * NN + n];
    float4 y = *((const float4*)(Y + (long)gw * HID) + lane);
    float* np = num + ((long)rc * NG + g) * HID + lane * 4;
    unsafeAtomicAdd(np + 0, y.x * m);
    unsafeAtomicAdd(np + 1, y.y * m);
    unsafeAtomicAdd(np + 2, y.z * m);
    unsafeAtomicAdd(np + 3, y.w * m);
    if (lane == 0) unsafeAtomicAdd(den + (long)rc * NG + g, m);
}

// out[((r*NG+g)*C + c)*128 + f] = num/(den+1e-7), with rc = r*4 + c.
__global__ void finalize_kernel(const float* __restrict__ num, const float* __restrict__ den,
                                float* __restrict__ out)
{
    int i = blockIdx.x * blockDim.x + threadIdx.x;
    if (i >= RCN * NG * HID) return;
    int f = i & 127;
    int rcg = i >> 7;
    int rc = rcg / NG, g = rcg - rc * NG;
    int r = rc >> 2, c = rc & 3;
    float v = num[i] / (den[rcg] + 1e-7f);
    out[(((long)r * NG + g) * 4 + c) * HID + f] = v;
}

extern "C" void kernel_launch(void* const* d_in, const int* in_sizes, int n_in,
                              void* d_out, int out_size, void* d_ws, size_t ws_size,
                              hipStream_t stream) {
    (void)in_sizes; (void)n_in; (void)out_size; (void)ws_size;
    const float* x_feat    = (const float*)d_in[0];
    const float* edge_attr = (const float*)d_in[1];
    const float* node_mask = (const float*)d_in[2];
    const float* edge_w    = (const float*)d_in[3];
    const float* W_atom    = (const float*)d_in[4];
    const float* b_atom    = (const float*)d_in[5];
    const float* eps       = (const float*)d_in[6];
    const float* W_bond    = (const float*)d_in[7];
    const float* b_bond    = (const float*)d_in[8];
    const float* W1        = (const float*)d_in[9];
    const float* b1        = (const float*)d_in[10];
    const float* W2        = (const float*)d_in[11];
    const float* b2        = (const float*)d_in[12];
    const float* Wm1       = (const float*)d_in[13];
    const float* bm1       = (const float*)d_in[14];
    const float* Wm2       = (const float*)d_in[15];
    const float* bm2       = (const float*)d_in[16];
    const int*   edge_idx  = (const int*)d_in[17];
    const int*   batch     = (const int*)d_in[18];
    const int* srcI = edge_idx;
    const int* dstI = edge_idx + NE;

    char* ws = (char*)d_ws;
    size_t off = 0;
    auto alloc = [&](size_t bytes) -> void* {
        void* p = ws + off;
        off = (off + bytes + 255) & ~(size_t)255;
        return p;
    };
    float*    xbuf   = (float*)alloc(sizeof(float) * (size_t)RCN * NN * HID);
    float*    aggbuf = (float*)alloc(sizeof(float) * (size_t)RCN * NN * HID);
    float*    eebuf  = (float*)alloc(sizeof(float) * (size_t)NE * HID);
    float*    num    = (float*)alloc(sizeof(float) * (size_t)RCN * NG * HID);
    float*    den    = (float*)alloc(sizeof(float) * (size_t)RCN * NG);  // contiguous after num
    _Float16* Wa16   = (_Float16*)alloc(sizeof(_Float16) * INF * HID);
    _Float16* W116   = (_Float16*)alloc(sizeof(_Float16) * NL * HID * HID);
    _Float16* W216   = (_Float16*)alloc(sizeof(_Float16) * NL * HID * HID);
    _Float16* Wm116  = (_Float16*)alloc(sizeof(_Float16) * HID * HID);
    _Float16* Wm216  = (_Float16*)alloc(sizeof(_Float16) * HID * HID);

    // Swizzle all GEMM weights into B-fragment-major f16 once.
    auto swz = [&](const float* s, _Float16* d, int K) {
        swizzle_w16_kernel<<<(K * HID + 255) / 256, 256, 0, stream>>>(s, d, K);
    };
    swz(W_atom, Wa16, INF);
    for (int l = 0; l < NL; ++l) {
        swz(W1 + l * HID * HID, W116 + l * HID * HID, HID);
        swz(W2 + l * HID * HID, W216 + l * HID * HID, HID);
    }
    swz(Wm1, Wm116, HID);
    swz(Wm2, Wm216, HID);

    atom_encoder_kernel<<<(NN / 16 + 7) / 8, 256, 0, stream>>>(x_feat, Wa16, b_atom, xbuf);

    const int mTotal = RCN * NN;   // 160000 rows = 1250 WGs * 8 waves * 16 rows exactly
    for (int l = 0; l < NL; ++l) {
        edge_emb_kernel<<<(NE * HID) / 256, 256, 0, stream>>>(
            edge_attr, W_bond + l * 16 * HID, b_bond + l * HID, eebuf);
        init_agg_kernel<<<(RCN * NN * HID / 4) / 256, 256, 0, stream>>>(xbuf, eps, l, aggbuf);
        msg_scatter_kernel<<<(RCN * NE) / 8, 256, 0, stream>>>(
            xbuf, eebuf, edge_w, srcI, dstI, aggbuf);
        fused_mlp_kernel<<<(mTotal / 16) / 8, 256, 0, stream>>>(
            aggbuf, W116 + l * HID * HID, b1 + l * HID,
            W216 + l * HID * HID, b2 + l * HID, xbuf, mTotal, 1);
    }
    // readout MLP (no trailing gelu), result into aggbuf
    fused_mlp_kernel<<<(mTotal / 16) / 8, 256, 0, stream>>>(
        xbuf, Wm116, bm1, Wm216, bm2, aggbuf, mTotal, 0);

    long nzero = (long)RCN * NG * HID + RCN * NG;   // num followed contiguously by den
    zero_kernel<<<(int)((nzero + 255) / 256), 256, 0, stream>>>(num, nzero);
    pool_scatter_kernel<<<(RCN * NN) / 8, 256, 0, stream>>>(aggbuf, node_mask, batch, num, den);
    finalize_kernel<<<(RCN * NG * HID) / 256, 256, 0, stream>>>(num, den, (float*)d_out);
}
